// Conv2d_mvm_14491219656718
// MI455X (gfx1250) — 8-bit matrix operand order repaired and checked by reading, compiled, not run on hardware
//
#include <hip/hip_runtime.h>
#include <hip/hip_bf16.h>
#include <stdint.h>


typedef int v8i __attribute__((ext_vector_type(8)));

#define K_TOT   576
#define KT_N    9
#define COUT    128
#define NT_N    8
#define P_TOT   2048
#define MT_N    128
#define A_DW    (MT_N * KT_N * 32 * 8)
#define B_DW    (KT_N * NT_N * 32 * 8)

__global__ void wq_kernel(const float* __restrict__ w,
                          uint32_t* __restrict__ bpos,
                          uint32_t* __restrict__ bneg) {
  int idx  = blockIdx.x * blockDim.x + threadIdx.x;
  int j    = idx & 7;
  int lane = (idx >> 3) & 31;
  int nt   = (idx >> 8) & 7;
  int kt   = idx >> 11;
  int n    = nt * 16 + (lane & 15);
  int half = lane >> 4;
  uint32_t p = 0u, q = 0u;
#pragma unroll
  for (int byte = 0; byte < 4; ++byte) {
    int k = kt * 64 + ((j & 4) ? 32 : 0) + half * 16 + (j & 3) * 4 + byte;
    float wv = w[n * K_TOT + k];
    int pp = (int)fminf(rintf(fmaxf( wv, 0.0f) * 64.0f), 255.0f);
    int qq = (int)fminf(rintf(fmaxf(-wv, 0.0f) * 64.0f), 255.0f);
    p |= ((uint32_t)pp) << (8 * byte);
    q |= ((uint32_t)qq) << (8 * byte);
  }
  bpos[idx] = p;
  bneg[idx] = q;
}

__global__ void iq_kernel(const float* __restrict__ x,
                          uint32_t* __restrict__ apack) {
  int idx  = blockIdx.x * blockDim.x + threadIdx.x;
  int j    = idx & 7;
  int lane = (idx >> 3) & 31;
  int t    = idx >> 8;
  int kt   = t % 9;
  int mt   = t / 9;
  int half = lane >> 4;
  int m    = mt * 16 + (lane & 15);
  int b    = m >> 8;
  int oh   = (m >> 4) & 15;
  int ow   = m & 15;
  uint32_t d = 0u;
#pragma unroll
  for (int byte = 0; byte < 4; ++byte) {
    int kl = ((j & 4) ? 32 : 0) + half * 16 + (j & 3) * 4 + byte;
    int l  = kt * 64 + kl;
    int cin = l / 9;
    int r   = l % 9;
    int ih  = oh + r / 3 - 1;
    int iw  = ow + r % 3 - 1;
    int q = 0;
    if (ih >= 0 && ih < 16 && iw >= 0 && iw < 16) {
      float v = x[((b * 64 + cin) * 16 + ih) * 16 + iw];
      q = (int)fminf(fmaxf(rintf(v * 64.0f), -128.0f), 127.0f);
    }
    d |= ((uint32_t)(q & 0xFF)) << (8 * byte);
  }
  apack[idx] = d;
}

__global__ void __launch_bounds__(256, 4)
gemm_kernel(const uint32_t* __restrict__ apack,
            const uint32_t* __restrict__ bpos,
            const uint32_t* __restrict__ bneg,
            float* __restrict__ out) {
  int lane = threadIdx.x & 31;
  int wave = threadIdx.x >> 5;
  int tile = blockIdx.x * 8 + wave;
  int mt   = tile >> 3;
  int nt   = tile & 7;

  const v8i* __restrict__ A  = ((const v8i*)apack) + (size_t)(mt * KT_N) * 32 + lane;
  const v8i* __restrict__ BP = ((const v8i*)bpos)  + (size_t)nt * 32 + lane;
  const v8i* __restrict__ BN = ((const v8i*)bneg)  + (size_t)nt * 32 + lane;

  v8i accP = {};
  v8i accN = {};
#pragma unroll
  for (int kt = 0; kt < KT_N; ++kt) {
    v8i a  = A[kt * 32];
    v8i bp = BP[kt * (NT_N * 32)];
    v8i bn = BN[kt * (NT_N * 32)];
    accP = __builtin_amdgcn_wmma_i32_16x16x64_iu8(
         true, a,  false, bp, accP, false, false);
    accN = __builtin_amdgcn_wmma_i32_16x16x64_iu8(
         true, a,  false, bn, accN, false, false);
  }

  int cout = nt * 16 + (lane & 15);
  int half = lane >> 4;
  int base = (((mt >> 4) * COUT + cout) * 16 + (mt & 15)) * 16 + half * 8;

  const float scale  = 1.0f / 4096.0f;
  const float acm_lo = -8.0f;
  const float acm_hi = 7.999755859375f;
  float f[8];
#pragma unroll
  for (int v = 0; v < 8; ++v) {
    int dfix = accP[v] - accN[v];
    f[v] = fminf(fmaxf((float)dfix * scale, acm_lo), acm_hi);
  }
  *(float4*)(out + base)     = make_float4(f[0], f[1], f[2], f[3]);
  *(float4*)(out + base + 4) = make_float4(f[4], f[5], f[6], f[7]);
}

extern "C" void kernel_launch(void* const* d_in, const int* in_sizes, int n_in,
                              void* d_out, int out_size, void* d_ws, size_t ws_size,
                              hipStream_t stream) {
  const float* x = (const float*)d_in[0];
  const float* w = (const float*)d_in[1];

  uint32_t* apack = (uint32_t*)d_ws;
  uint32_t* bpos  = apack + A_DW;
  uint32_t* bneg  = bpos + B_DW;

  wq_kernel<<<B_DW / 256, 256, 0, stream>>>(w, bpos, bneg);
  iq_kernel<<<A_DW / 256, 256, 0, stream>>>(x, apack);
  gemm_kernel<<<MT_N * NT_N / 8, 256, 0, stream>>>(apack, bpos, bneg, (float*)d_out);
}
